// Model_33311766347793
// MI455X (gfx1250) — compile-verified, matrix operand order checked by reading, not run on hardware
//
#include <hip/hip_runtime.h>
#include <hip/hip_bf16.h>

// ---------------------------------------------------------------------------
// CDNA5 (gfx1250) implementation of the GAT+GIP+Laplacian+MLP pipeline.
// Core compute: bf16 WMMA GEMM (V_WMMA_F32_16X16X32_BF16), f32 accumulate,
// async global->LDS staging of the A tile (ASYNCcnt), vectorized b128
// LDS fragment loads.
// ---------------------------------------------------------------------------

typedef __attribute__((ext_vector_type(16))) __bf16          v16bf;
typedef __attribute__((ext_vector_type(16))) unsigned short  v16us;
typedef __attribute__((ext_vector_type(8)))  unsigned short  v8us;
typedef __attribute__((ext_vector_type(4)))  unsigned short  v4us;
typedef __attribute__((ext_vector_type(4)))  float           v4f;
typedef __attribute__((ext_vector_type(8)))  float           v8f;

#define BM 128
#define BN 64
#define BK 32

__device__ __forceinline__ unsigned short f2bfbits(float f) {
  union { float f; unsigned u; } v; v.f = f;
  unsigned u = v.u;
  unsigned rnd = 0x7FFFu + ((u >> 16) & 1u);   // round-to-nearest-even
  return (unsigned short)((u + rnd) >> 16);
}

__device__ __forceinline__ float epi(float v, float b, int act) {
  v += b;
  if (act == 1) v = v > 0.f ? v : 0.f;                 // relu
  else if (act == 2) v = 1.f / (1.f + __expf(-v));     // sigmoid
  return v;
}

__device__ __forceinline__ v16bf frag_from_lds(const unsigned short* row, int half) {
  // fragment K pattern (ISA 7.12.2, 16-bit 16x32): two contiguous 8-elem runs
  v8us lo = *(const v8us*)(row + half * 8);
  v8us hi = *(const v8us*)(row + 16 + half * 8);
  v16us full = __builtin_shufflevector(lo, hi, 0, 1, 2, 3, 4, 5, 6, 7,
                                       8, 9, 10, 11, 12, 13, 14, 15);
  return __builtin_bit_cast(v16bf, full);
}

// Generic-stride GEMM: C[m,n] = act( sum_k A[m*lar+k*lac]*B[k*lbr+n*lbc] + bias[n] )
// Requires K % 32 == 0 (true for every call site in this pipeline).
__global__ __launch_bounds__(256) void gemm_bf16_wmma(
    const float* __restrict__ A, long long lar, long long lac,
    const float* __restrict__ B, long long lbr, long long lbc,
    const float* __restrict__ bias,
    float* __restrict__ C, long long ldc,
    int M, int Nn, int K, int act)
{
  __shared__ float          Af32[BM * BK];   // async staging, 16 KB
  __shared__ unsigned short As[BM * BK];     // bf16 A tile [m][k], 8 KB
  __shared__ unsigned short Bt[BN * BK];     // bf16 B tile transposed [n][k], 4 KB

  const int tid  = threadIdx.x;
  const int wave = tid >> 5;
  const int lane = tid & 31;
  const int l15  = lane & 15;
  const int half = lane >> 4;

  const int m0 = blockIdx.y * BM;
  const int n0 = blockIdx.x * BN;

  // 8 waves: 4 row-groups x 2 col-groups, each wave owns a 32x32 tile (2x2 WMMA)
  const int wm = (wave & 3) * 32;
  const int wn = (wave >> 2) * 32;

  v8f acc[2][2];
  #pragma unroll
  for (int a = 0; a < 2; ++a)
    #pragma unroll
    for (int b = 0; b < 2; ++b)
      #pragma unroll
      for (int i = 0; i < 8; ++i) acc[a][b][i] = 0.f;

  for (int k0 = 0; k0 < K; k0 += BK) {
    // ---- A tile: 128x32 f32 staged via async global->LDS, then f32->bf16
    if (lac == 1) {
      #pragma unroll
      for (int i = 0; i < 4; ++i) {
        int idx = i * 256 + tid;              // float4 index (0..1023)
        int m = idx >> 3, kq = (idx & 7) * 4;
        int gm = m0 + m;
        if (gm < M) {
          unsigned ldsoff = (unsigned)(unsigned long long)(const void*)&Af32[idx * 4];
          unsigned long long ga =
              (unsigned long long)(const void*)&A[(long long)gm * lar + (long long)(k0 + kq)];
          asm volatile("global_load_async_to_lds_b128 %0, %1, off"
                       :: "v"(ldsoff), "v"(ga) : "memory");
        } else {
          v4f z; z[0] = z[1] = z[2] = z[3] = 0.f;
          *(v4f*)&Af32[idx * 4] = z;
        }
      }
      asm volatile("s_wait_asynccnt 0x0" ::: "memory");
      // convert the thread's own staged quads (thread-private region of Af32)
      #pragma unroll
      for (int i = 0; i < 4; ++i) {
        int idx = i * 256 + tid;
        v4f v = *(const v4f*)&Af32[idx * 4];
        v4us h;
        #pragma unroll
        for (int j = 0; j < 4; ++j) h[j] = f2bfbits(v[j]);
        *(v4us*)&As[idx * 4] = h;
      }
    } else {
      #pragma unroll
      for (int i = 0; i < 16; ++i) {
        int idx = i * 256 + tid;
        int m = idx >> 5, k = idx & 31;
        float v = 0.f;
        int gm = m0 + m;
        if (gm < M) v = A[(long long)gm * lar + (long long)(k0 + k) * lac];
        As[m * BK + k] = f2bfbits(v);
      }
    }

    // ---- B tile -> Bt[n][k]
    if (lbc == 1) {
      // row-major B: contiguous along n; prefetch next K tile while here
      if (k0 + BK < K) {
        int pn = n0 + (tid & 63);
        if (pn < Nn)
          __builtin_prefetch(&B[(long long)(k0 + BK) * lbr + pn], 0, 1);
      }
      #pragma unroll
      for (int i = 0; i < 2; ++i) {
        int idx = i * 256 + tid;              // 0..511
        int k = idx >> 4, nq = (idx & 15) * 4;
        long long grow = (long long)(k0 + k) * lbr;
        #pragma unroll
        for (int j = 0; j < 4; ++j) {
          int gn = n0 + nq + j;
          float v = (gn < Nn) ? B[grow + gn] : 0.f;
          Bt[(nq + j) * BK + k] = f2bfbits(v);
        }
      }
    } else if (lbr == 1) {
      // transposed B: contiguous along k
      #pragma unroll
      for (int i = 0; i < 2; ++i) {
        int idx = i * 256 + tid;
        int n = idx >> 3, kq = (idx & 7) * 4;
        int gn = n0 + n;
        v4us h;
        #pragma unroll
        for (int j = 0; j < 4; ++j) {
          float v = (gn < Nn) ? B[(long long)(k0 + kq + j) + (long long)gn * lbc] : 0.f;
          h[j] = f2bfbits(v);
        }
        *(v4us*)&Bt[n * BK + kq] = h;
      }
    } else {
      #pragma unroll
      for (int i = 0; i < 8; ++i) {
        int idx = i * 256 + tid;
        int n = idx >> 5, k = idx & 31;
        int gn = n0 + n;
        float v = (gn < Nn) ? B[(long long)(k0 + k) * lbr + (long long)gn * lbc] : 0.f;
        Bt[n * BK + k] = f2bfbits(v);
      }
    }
    __syncthreads();

    // ---- fragments (two ds_load_b128 each) + 4 WMMAs
    v16bf afr[2], bfr[2];
    #pragma unroll
    for (int t = 0; t < 2; ++t) {
      afr[t] = frag_from_lds(&As[(wm + t * 16 + l15) * BK], half);
      bfr[t] = frag_from_lds(&Bt[(wn + t * 16 + l15) * BK], half);
    }
    acc[0][0] = __builtin_amdgcn_wmma_f32_16x16x32_bf16(false, afr[0], false, bfr[0],
                                                        (short)0, acc[0][0], false, false);
    acc[0][1] = __builtin_amdgcn_wmma_f32_16x16x32_bf16(false, afr[0], false, bfr[1],
                                                        (short)0, acc[0][1], false, false);
    acc[1][0] = __builtin_amdgcn_wmma_f32_16x16x32_bf16(false, afr[1], false, bfr[0],
                                                        (short)0, acc[1][0], false, false);
    acc[1][1] = __builtin_amdgcn_wmma_f32_16x16x32_bf16(false, afr[1], false, bfr[1],
                                                        (short)0, acc[1][1], false, false);
    __syncthreads();
  }

  // C/D layout: lane l15 = column; VGPR r -> row r + 8*half.
  #pragma unroll
  for (int tr = 0; tr < 2; ++tr) {
    #pragma unroll
    for (int tc = 0; tc < 2; ++tc) {
      int cn = n0 + wn + tc * 16 + l15;
      if (cn >= Nn) continue;
      float bb = bias ? bias[cn] : 0.f;
      #pragma unroll
      for (int r = 0; r < 8; ++r) {
        int cm = m0 + wm + tr * 16 + r + half * 8;
        if (cm < M) C[(long long)cm * ldc + cn] = epi(acc[tr][tc][r], bb, act);
      }
    }
  }
}

// ------------------------------- streaming kernels -------------------------

__global__ void fill_f32(float* p, float v, long long n) {
  long long i = (long long)blockIdx.x * 256 + threadIdx.x;
  if (i < n) p[i] = v;
}

__global__ void row_dots(const float* __restrict__ h,
                         const float* __restrict__ a_s,
                         const float* __restrict__ a_d,
                         float* ss, float* sd, int F, int NN) {
  int i = blockIdx.x * 256 + threadIdx.x;
  if (i >= NN) return;
  float s = 0.f, d = 0.f;
  const float* row = h + (long long)i * F;
  for (int f = 0; f < F; ++f) { float v = row[f]; s += v * a_s[f]; d += v * a_d[f]; }
  ss[i] = s; sd[i] = d;
}

__device__ __forceinline__ void atomicMaxF(float* addr, float val) {
  unsigned* ua = (unsigned*)addr;
  unsigned old = *ua;
  while (__uint_as_float(old) < val) {
    unsigned assumed = old;
    old = atomicCAS(ua, assumed, __float_as_uint(val));
    if (old == assumed) break;
  }
}
__device__ __forceinline__ void atomicMinF(float* addr, float val) {
  unsigned* ua = (unsigned*)addr;
  unsigned old = *ua;
  while (__uint_as_float(old) > val) {
    unsigned assumed = old;
    old = atomicCAS(ua, assumed, __float_as_uint(val));
    if (old == assumed) break;
  }
}

__global__ void edge_logit_max(const int* __restrict__ ei, int E_, int NN,
                               const float* __restrict__ ss, const float* __restrict__ sd,
                               float* logit, float* mseg) {
  long long e = (long long)blockIdx.x * 256 + threadIdx.x;
  long long E2 = (long long)E_ + NN;
  if (e >= E2) return;
  int s, dn;
  if (e < E_) { s = ei[e]; dn = ei[E_ + e]; } else { s = dn = (int)(e - E_); }
  float l = ss[s] + sd[dn];
  l = l > 0.f ? l : 0.2f * l;                 // leaky_relu, NEG = 0.2
  logit[e] = l;
  atomicMaxF(&mseg[dn], l);
}

__global__ void edge_exp_sum(const int* __restrict__ ei, int E_, int NN,
                             float* logit, const float* __restrict__ mseg, float* sseg) {
  long long e = (long long)blockIdx.x * 256 + threadIdx.x;
  long long E2 = (long long)E_ + NN;
  if (e >= E2) return;
  int dn = (e < E_) ? ei[E_ + e] : (int)(e - E_);
  float ex = __expf(logit[e] - mseg[dn]);
  logit[e] = ex;
  atomicAdd(&sseg[dn], ex);
}

__global__ void edge_scatter(const int* __restrict__ ei, int E_, int NN,
                             const float* __restrict__ ex, const float* __restrict__ sseg,
                             const float* __restrict__ h, float* agg, int F) {
  long long e = blockIdx.x;
  int f = blockIdx.y * 256 + threadIdx.x;
  if (f >= F) return;
  int s, dn;
  if (e < E_) { s = ei[e]; dn = ei[E_ + e]; } else { s = dn = (int)(e - E_); }
  float a = ex[e] / sseg[dn];
  atomicAdd(&agg[(long long)dn * F + f], a * h[(long long)s * F + f]);
}

__global__ void bias_relu(float* X, const float* __restrict__ b, int F, long long n) {
  long long i = (long long)blockIdx.x * 256 + threadIdx.x;
  if (i >= n) return;
  int f = (int)(i % F);
  float v = X[i] + b[f];
  X[i] = v > 0.f ? v : 0.f;
}

__global__ void minmax_kernel(const float* __restrict__ y, long long n, float* mn, float* mx) {
  __shared__ float smn[256], smx[256];
  float lmn = 3.4e38f, lmx = -3.4e38f;
  for (long long i = (long long)blockIdx.x * 256 + threadIdx.x; i < n;
       i += (long long)gridDim.x * 256) {
    float v = y[i]; lmn = fminf(lmn, v); lmx = fmaxf(lmx, v);
  }
  smn[threadIdx.x] = lmn; smx[threadIdx.x] = lmx;
  __syncthreads();
  for (int o = 128; o > 0; o >>= 1) {
    if (threadIdx.x < o) {
      smn[threadIdx.x] = fminf(smn[threadIdx.x], smn[threadIdx.x + o]);
      smx[threadIdx.x] = fmaxf(smx[threadIdx.x], smx[threadIdx.x + o]);
    }
    __syncthreads();
  }
  if (threadIdx.x == 0) { atomicMinF(mn, smn[0]); atomicMaxF(mx, smx[0]); }
}

__global__ void normalize_k(const float* __restrict__ y, float* yn, long long n,
                            const float* mnp, const float* mxp) {
  long long i = (long long)blockIdx.x * 256 + threadIdx.x;
  if (i >= n) return;
  float mn = *mnp;
  float inv = 1.f / (*mxp - mn + 1e-12f);
  yn[i] = (y[i] - mn) * inv;
}

__global__ void diag_sum(const float* __restrict__ k, int Nr, float* out) {
  __shared__ float s[256];
  float v = 0.f;
  for (int i = blockIdx.x * 256 + threadIdx.x; i < Nr; i += gridDim.x * 256)
    v += k[(long long)i * Nr + i];
  s[threadIdx.x] = v;
  __syncthreads();
  for (int o = 128; o > 0; o >>= 1) {
    if (threadIdx.x < o) s[threadIdx.x] += s[threadIdx.x + o];
    __syncthreads();
  }
  if (threadIdx.x == 0) atomicAdd(out, s[0]);
}

__global__ void gip_accum(float* outk, const float* __restrict__ km, int Nr,
                          const float* dsum, const float* att, float gamma) {
  long long idx = (long long)blockIdx.x * 256 + threadIdx.x;
  long long nn = (long long)Nr * Nr;
  if (idx >= nn) return;
  int i = (int)(idx / Nr), j = (int)(idx % Nr);
  float inv = (float)Nr / (*dsum);        // 1/meanDiag
  float kij = km[idx];
  float kii = km[(long long)i * Nr + i];
  float kjj = km[(long long)j * Nr + j];
  outk[idx] += (*att) * __expf(-gamma * (kii + kjj - 2.f * kij) * inv);
}

__global__ void scale_copy(float* dst, const float* __restrict__ src,
                           const float* att, long long n) {
  long long i = (long long)blockIdx.x * 256 + threadIdx.x;
  if (i < n) dst[i] = (*att) * src[i];
}

__global__ void rowsum_k(const float* __restrict__ k, int Nr, float* r) {
  int i = blockIdx.x * 256 + threadIdx.x;
  if (i >= Nr) return;
  const float* row = k + (long long)i * Nr;
  float s = 0.f;
  for (int j = 0; j < Nr; ++j) s += row[j];
  r[i] = s;
}

// bf[i,j] = ( r[i]*alpha[i,j] - G[i,j] > 0 ) ? 1 : 0   (lap(K)@A = r_i*A - K@A)
__global__ void lap_bin(const float* __restrict__ r, const float* __restrict__ alpha,
                        const float* __restrict__ G, float* bf, int Nr, int Nc) {
  long long idx = (long long)blockIdx.x * 256 + threadIdx.x;
  long long nn = (long long)Nr * Nc;
  if (idx >= nn) return;
  int i = (int)(idx / Nc);
  float v = r[i] * alpha[idx] - G[idx];
  bf[idx] = v > 0.f ? 1.f : 0.f;
}

// ------------------------------- host side ---------------------------------

static inline unsigned gblocks(long long n) { return (unsigned)((n + 255) / 256); }

static void gemm(hipStream_t st, const float* A, long long lar, long long lac,
                 const float* B, long long lbr, long long lbc,
                 const float* bias, float* C, long long ldc,
                 int M, int Nn, int K, int act) {
  dim3 g((Nn + BN - 1) / BN, (M + BM - 1) / BM);
  gemm_bf16_wmma<<<g, 256, 0, st>>>(A, lar, lac, B, lbr, lbc, bias, C, ldc, M, Nn, K, act);
}

static void run_gat(hipStream_t st, const float* X, int K, const float* W, int F,
                    const float* a_s, const float* a_d, const float* bias,
                    const int* ei, float* hbuf, float* H,
                    float* ss, float* sd, float* mseg, float* sseg, float* ex) {
  const int NN = 8192, E_ = 131072;
  const long long E2 = (long long)E_ + NN;
  gemm(st, X, K, 1, W, F, 1, nullptr, hbuf, F, NN, F, K, 0);
  row_dots<<<gblocks(NN), 256, 0, st>>>(hbuf, a_s, a_d, ss, sd, F, NN);
  fill_f32<<<gblocks(NN), 256, 0, st>>>(mseg, -3.4e38f, NN);
  fill_f32<<<gblocks(NN), 256, 0, st>>>(sseg, 0.f, NN);
  long long nH = (long long)NN * F;
  fill_f32<<<gblocks(nH), 256, 0, st>>>(H, 0.f, nH);
  edge_logit_max<<<gblocks(E2), 256, 0, st>>>(ei, E_, NN, ss, sd, ex, mseg);
  edge_exp_sum<<<gblocks(E2), 256, 0, st>>>(ei, E_, NN, ex, mseg, sseg);
  dim3 g((unsigned)E2, (F + 255) / 256);
  edge_scatter<<<g, 256, 0, st>>>(ei, E_, NN, ex, sseg, hbuf, H, F);
  bias_relu<<<gblocks(nH), 256, 0, st>>>(H, bias, F, nH);
}

static void run_gip(hipStream_t st, const float* Hl, int F, int rowOff, int Nr,
                    float gamma, const float* attp, float* outk,
                    float* yn, float* kbuf, float* scal) {
  const float* y = Hl + (long long)rowOff * F;
  long long n = (long long)Nr * F;
  fill_f32<<<1, 1, 0, st>>>(scal + 0, 3.4e38f, 1);
  fill_f32<<<1, 1, 0, st>>>(scal + 1, -3.4e38f, 1);
  fill_f32<<<1, 1, 0, st>>>(scal + 2, 0.f, 1);
  unsigned mb = gblocks(n); if (mb > 1024) mb = 1024;
  minmax_kernel<<<mb, 256, 0, st>>>(y, n, scal + 0, scal + 1);
  normalize_k<<<gblocks(n), 256, 0, st>>>(y, yn, n, scal + 0, scal + 1);
  gemm(st, yn, F, 1, yn, 1, F, nullptr, kbuf, Nr, Nr, Nr, F, 0);
  diag_sum<<<64, 256, 0, st>>>(kbuf, Nr, scal + 2);
  long long nn = (long long)Nr * Nr;
  gip_accum<<<gblocks(nn), 256, 0, st>>>(outk, kbuf, Nr, scal + 2, attp, gamma);
}

static void run_branch(hipStream_t st, const float* kk, int Nr,
                       const float* alpha, int Nc,
                       const float* cmW, const float* cmb,
                       const float* fc1W, const float* fc1b,
                       const float* fc2W, const float* fc2b,
                       const float* fc3W, const float* fc3b,
                       const float* fc4W, const float* fc4b,
                       float* r, float* G, float* bfm, float* c,
                       float* t1, float* t2, float* t3, float* o) {
  rowsum_k<<<gblocks(Nr), 256, 0, st>>>(kk, Nr, r);
  gemm(st, kk, Nr, 1, alpha, Nc, 1, nullptr, G, Nc, Nr, Nc, Nr, 0);
  long long nn = (long long)Nr * Nc;
  lap_bin<<<gblocks(nn), 256, 0, st>>>(r, alpha, G, bfm, Nr, Nc);
  gemm(st, bfm, Nc, 1, cmW, 1, Nc, cmb, c, Nc, Nr, Nc, Nc, 1);       // relu
  gemm(st, c, Nc, 1, fc1W, 1, Nc, fc1b, t1, 256, Nr, 256, Nc, 1);    // relu
  gemm(st, t1, 256, 1, fc2W, 1, 256, fc2b, t2, 128, Nr, 128, 256, 1);
  gemm(st, t2, 128, 1, fc3W, 1, 128, fc3b, t3, 64, Nr, 64, 128, 1);
  gemm(st, t3, 64, 1, fc4W, 1, 64, fc4b, o, 32, Nr, 32, 64, 2);      // sigmoid
}

extern "C" void kernel_launch(void* const* d_in, const int* in_sizes, int n_in,
                              void* d_out, int out_size, void* d_ws, size_t ws_size,
                              hipStream_t stream) {
  (void)in_sizes; (void)n_in; (void)out_size; (void)ws_size;
  const float* x        = (const float*)d_in[0];
  const int*   ei       = (const int*)d_in[1];
  const float* pro_sim  = (const float*)d_in[2];
  const float* drug_sim = (const float*)d_in[3];
  const float* att_m    = (const float*)d_in[4];
  const float* att_d    = (const float*)d_in[5];
  const float* W1  = (const float*)d_in[6];
  const float* as1 = (const float*)d_in[7];
  const float* ad1 = (const float*)d_in[8];
  const float* b1  = (const float*)d_in[9];
  const float* W2  = (const float*)d_in[10];
  const float* as2 = (const float*)d_in[11];
  const float* ad2 = (const float*)d_in[12];
  const float* b2  = (const float*)d_in[13];
  const float* W3  = (const float*)d_in[14];
  const float* as3 = (const float*)d_in[15];
  const float* ad3 = (const float*)d_in[16];
  const float* b3  = (const float*)d_in[17];
  const float* alpha1 = (const float*)d_in[18];
  const float* alpha2 = (const float*)d_in[19];
  const float* cm1_W  = (const float*)d_in[20];
  const float* cm1_b  = (const float*)d_in[21];
  const float* cm2_W  = (const float*)d_in[22];
  const float* cm2_b  = (const float*)d_in[23];
  const float* fc11_W = (const float*)d_in[24];
  const float* fc11_b = (const float*)d_in[25];
  const float* fc12_W = (const float*)d_in[26];
  const float* fc12_b = (const float*)d_in[27];
  const float* fc2_W  = (const float*)d_in[28];
  const float* fc2_b  = (const float*)d_in[29];
  const float* fc3_W  = (const float*)d_in[30];
  const float* fc3_b  = (const float*)d_in[31];
  const float* fc4_W  = (const float*)d_in[32];
  const float* fc4_b  = (const float*)d_in[33];
  const float* Wm     = (const float*)d_in[34];
  float* out = (float*)d_out;

  // Workspace bump allocator (peak ~380 MB).
  char* base = (char*)d_ws;
  size_t off = 0;
  auto alloc = [&](size_t bytes) -> float* {
    float* p = (float*)(base + off);
    off += (bytes + 255) & ~(size_t)255;
    return p;
  };
  float* H1    = alloc(8192ULL * 512 * 4);
  float* H2    = alloc(8192ULL * 256 * 4);
  float* H3    = alloc(8192ULL * 128 * 4);
  float* prok  = alloc(4096ULL * 4096 * 4);
  float* drugk = alloc(4096ULL * 4096 * 4);
  float* S1    = alloc(4096ULL * 4096 * 4);   // hbuf / kbuf / G
  float* S2    = alloc(4096ULL * 4096 * 4);   // binarized features
  float* S3    = alloc(4096ULL * 4096 * 4);   // contrastive features
  float* yn    = alloc(4096ULL * 512 * 4);
  float* t1    = alloc(4096ULL * 256 * 4);
  float* t2    = alloc(4096ULL * 128 * 4);
  float* t3    = alloc(4096ULL * 64 * 4);
  float* o1    = alloc(4096ULL * 32 * 4);
  float* o2    = alloc(4096ULL * 32 * 4);
  float* tmpm  = alloc(4096ULL * 32 * 4);
  float* ss    = alloc(8192 * 4);
  float* sd    = alloc(8192 * 4);
  float* mseg  = alloc(8192 * 4);
  float* sseg  = alloc(8192 * 4);
  float* ex    = alloc((size_t)(131072 + 8192) * 4);
  float* r     = alloc(4096 * 4);
  float* scal  = alloc(256);

  // ---- 3 GAT layers (S1 doubles as the pre-aggregation projection buffer)
  run_gat(stream, x,  8192, W1, 512, as1, ad1, b1, ei, S1, H1, ss, sd, mseg, sseg, ex);
  run_gat(stream, H1, 512,  W2, 256, as2, ad2, b2, ei, S1, H2, ss, sd, mseg, sseg, ex);
  run_gat(stream, H2, 256,  W3, 128, as3, ad3, b3, ei, S1, H3, ss, sd, mseg, sseg, ex);

  // ---- GIP kernel fusion: pro_k / drug_k
  const long long pp = 4096LL * 4096;
  const float G123 = 0.03125f;
  scale_copy<<<gblocks(pp), 256, 0, stream>>>(prok, pro_sim, att_m + 3, pp);
  run_gip(stream, H1, 512, 0, 4096, G123, att_m + 0, prok, yn, S1, scal);
  run_gip(stream, H2, 256, 0, 4096, G123, att_m + 1, prok, yn, S1, scal);
  run_gip(stream, H3, 128, 0, 4096, G123, att_m + 2, prok, yn, S1, scal);
  scale_copy<<<gblocks(pp), 256, 0, stream>>>(drugk, drug_sim, att_d + 3, pp);
  run_gip(stream, H1, 512, 4096, 4096, G123, att_d + 0, drugk, yn, S1, scal);
  run_gip(stream, H2, 256, 4096, 4096, G123, att_d + 1, drugk, yn, S1, scal);
  run_gip(stream, H3, 128, 4096, 4096, G123, att_d + 2, drugk, yn, S1, scal);

  // ---- Laplacian -> binarize -> contrastive -> MLP, both branches
  run_branch(stream, prok, 4096, alpha1, 4096, cm1_W, cm1_b, fc11_W, fc11_b,
             fc2_W, fc2_b, fc3_W, fc3_b, fc4_W, fc4_b,
             r, S1, S2, S3, t1, t2, t3, o1);
  run_branch(stream, drugk, 4096, alpha2, 4096, cm2_W, cm2_b, fc12_W, fc12_b,
             fc2_W, fc2_b, fc3_W, fc3_b, fc4_W, fc4_b,
             r, S1, S2, S3, t1, t2, t3, o2);

  // ---- final: (o1 @ Wm) @ o2^T -> [P, D]
  gemm(stream, o1, 32, 1, Wm, 32, 1, nullptr, tmpm, 32, 4096, 32, 32, 0);
  gemm(stream, tmpm, 32, 1, o2, 1, 32, nullptr, out, 4096, 4096, 4096, 32, 0);
}
